// S6Layer_61881888800885
// MI455X (gfx1250) — matrix operand order repaired and checked by reading, not compiled, not run on hardware
//
#include <hip/hip_runtime.h>
#include <hip/hip_bf16.h>

// ---------------------------------------------------------------------------
// Mamba S6 layer for gfx1250 (MI455X).
// Big GEMMs: bf16x3 split (fp32-class accuracy) on v_wmma_f32_16x16x32_bf16.
// xp GEMM:   exact fp32 on v_wmma_f32_16x16x4_f32.
// Scan:      fused channel-parallel recurrence.
// ---------------------------------------------------------------------------

typedef __attribute__((ext_vector_type(2)))  float    v2f;
typedef __attribute__((ext_vector_type(8)))  float    v8f;
typedef __attribute__((ext_vector_type(16))) __bf16   v16bf;
typedef __attribute__((ext_vector_type(4)))  unsigned v4u;

static constexpr int BATCH   = 2;
static constexpr int LSEQ    = 1024;
static constexpr int DMODEL  = 1024;
static constexpr int DINNER  = 2048;
static constexpr int DSTATE  = 16;
static constexpr int M_ROWS  = BATCH * LSEQ;      // 2048
static constexpr int NXP     = 2 * DSTATE + 1;    // 33

// ----------------------------- bf16 helpers --------------------------------
static __device__ __forceinline__ unsigned short f2bf(float f) {
  unsigned u = __builtin_bit_cast(unsigned, f);
  u += 0x7FFFu + ((u >> 16) & 1u);                // round-to-nearest-even
  return (unsigned short)(u >> 16);
}
static __device__ __forceinline__ float bf2f(unsigned short h) {
  unsigned u = ((unsigned)h) << 16;
  return __builtin_bit_cast(float, u);
}

struct FragB32 { unsigned u[8]; };                // 16 bf16 = 32B per lane

static __device__ __forceinline__ v8f wmma_bf16(const FragB32& a,
                                                const FragB32& b, v8f c) {
  return __builtin_amdgcn_wmma_f32_16x16x32_bf16(
      false, __builtin_bit_cast(v16bf, a), false, __builtin_bit_cast(v16bf, b),
      (short)0, c, false, false);
}
static __device__ __forceinline__ v8f wmma_f32(v2f a, v2f b, v8f c) {
  return __builtin_amdgcn_wmma_f32_16x16x4_f32(
      false, a, false, b, (short)0, c, false, false);
}

// ---------------------------------------------------------------------------
// Split fp32 -> (hi, lo) bf16 arrays.  a ~= hi + lo with ~16 mantissa bits.
// ---------------------------------------------------------------------------
__global__ __launch_bounds__(256)
void split_bf16_kernel(const float* __restrict__ src,
                       unsigned short* __restrict__ hi,
                       unsigned short* __restrict__ lo, int n) {
  const int i = blockIdx.x * 256 + threadIdx.x;
  if (i >= n) return;
  const float f = src[i];
  const unsigned short h = f2bf(f);
  hi[i] = h;
  lo[i] = f2bf(f - bf2f(h));
}

// ---------------------------------------------------------------------------
// bf16x3 GEMM: D[M x N] = A[M x K] * W[N x K]^T, A/W pre-split into hi/lo
// bf16 row-major arrays.  block = 128 thr = 4 waves (2x2), block tile
// 32(M) x 128(N), wave tile 16 x 64 (4 subtiles), K stepped by 32.
// grid = (N/128, M/32).
// Frag layouts (wave32):
//   A 16x32 bf16 : lane m = lane&15, ka = 16*(lane>>4);
//                  u[0..7] <- K [ka, ka+16)
//   B 32x16 bf16 : lane n = lane&15, kb = 16*(lane>>4);
//                  u[0..7] <- K [kb, kb+16)
// ---------------------------------------------------------------------------
__global__ __launch_bounds__(128)
void gemm_bf16x3(const unsigned short* __restrict__ Ahi,
                 const unsigned short* __restrict__ Alo,
                 const unsigned short* __restrict__ Whi,
                 const unsigned short* __restrict__ Wlo,
                 float* __restrict__ D, int K, int ldD) {
  const int lane = threadIdx.x & 31;
  const int wave = threadIdx.x >> 5;
  const int l15  = lane & 15;
  const int half = lane >> 4;

  const int arow  = blockIdx.y * 32 + (wave >> 1) * 16 + l15;
  const int nbase = blockIdx.x * 128 + (wave & 1) * 64 + l15;

  const size_t aoff = (size_t)arow * K + 16 * half;    // + k0 (+8 for u[4..7])
  const size_t boff0 = (size_t)(nbase +  0) * K + 16 * half;
  const size_t boff1 = (size_t)(nbase + 16) * K + 16 * half;
  const size_t boff2 = (size_t)(nbase + 32) * K + 16 * half;
  const size_t boff3 = (size_t)(nbase + 48) * K + 16 * half;

  v8f acc0 = {}, acc1 = {}, acc2 = {}, acc3 = {};

  for (int k0 = 0; k0 < K; k0 += 32) {
    // ---- batch all fragment loads for this 32-K slab -----------------------
    FragB32 ah, al, bh0, bl0, bh1, bl1, bh2, bl2, bh3, bl3;
    *(v4u*)&ah.u[0] = *(const v4u*)(Ahi + aoff + k0);
    *(v4u*)&ah.u[4] = *(const v4u*)(Ahi + aoff + k0 + 8);
    *(v4u*)&al.u[0] = *(const v4u*)(Alo + aoff + k0);
    *(v4u*)&al.u[4] = *(const v4u*)(Alo + aoff + k0 + 8);

    *(v4u*)&bh0.u[0] = *(const v4u*)(Whi + boff0 + k0);
    *(v4u*)&bh0.u[4] = *(const v4u*)(Whi + boff0 + k0 + 8);
    *(v4u*)&bl0.u[0] = *(const v4u*)(Wlo + boff0 + k0);
    *(v4u*)&bl0.u[4] = *(const v4u*)(Wlo + boff0 + k0 + 8);
    *(v4u*)&bh1.u[0] = *(const v4u*)(Whi + boff1 + k0);
    *(v4u*)&bh1.u[4] = *(const v4u*)(Whi + boff1 + k0 + 8);
    *(v4u*)&bl1.u[0] = *(const v4u*)(Wlo + boff1 + k0);
    *(v4u*)&bl1.u[4] = *(const v4u*)(Wlo + boff1 + k0 + 8);
    *(v4u*)&bh2.u[0] = *(const v4u*)(Whi + boff2 + k0);
    *(v4u*)&bh2.u[4] = *(const v4u*)(Whi + boff2 + k0 + 8);
    *(v4u*)&bl2.u[0] = *(const v4u*)(Wlo + boff2 + k0);
    *(v4u*)&bl2.u[4] = *(const v4u*)(Wlo + boff2 + k0 + 8);
    *(v4u*)&bh3.u[0] = *(const v4u*)(Whi + boff3 + k0);
    *(v4u*)&bh3.u[4] = *(const v4u*)(Whi + boff3 + k0 + 8);
    *(v4u*)&bl3.u[0] = *(const v4u*)(Wlo + boff3 + k0);
    *(v4u*)&bl3.u[4] = *(const v4u*)(Wlo + boff3 + k0 + 8);

    // prefetch next slab (lowers to global_prefetch_b8)
    if (k0 + 32 < K) {
      __builtin_prefetch(Ahi + aoff + k0 + 32, 0, 1);
      __builtin_prefetch(Whi + boff0 + k0 + 32, 0, 1);
      __builtin_prefetch(Whi + boff1 + k0 + 32, 0, 1);
      __builtin_prefetch(Whi + boff2 + k0 + 32, 0, 1);
      __builtin_prefetch(Whi + boff3 + k0 + 32, 0, 1);
    }

    // ---- 12 back-to-back WMMAs: hi*hi + hi*lo + lo*hi ----------------------
    acc0 = wmma_bf16(ah, bh0, acc0);
    acc1 = wmma_bf16(ah, bh1, acc1);
    acc2 = wmma_bf16(ah, bh2, acc2);
    acc3 = wmma_bf16(ah, bh3, acc3);
    acc0 = wmma_bf16(ah, bl0, acc0);
    acc1 = wmma_bf16(ah, bl1, acc1);
    acc2 = wmma_bf16(ah, bl2, acc2);
    acc3 = wmma_bf16(ah, bl3, acc3);
    acc0 = wmma_bf16(al, bh0, acc0);
    acc1 = wmma_bf16(al, bh1, acc1);
    acc2 = wmma_bf16(al, bh2, acc2);
    acc3 = wmma_bf16(al, bh3, acc3);
  }

  const int srow0 = blockIdx.y * 32 + (wave >> 1) * 16 + half * 8;
  #pragma unroll
  for (int v = 0; v < 8; ++v) {
    float* drow = D + (size_t)(srow0 + v) * ldD;
    drow[nbase +  0] = acc0[v];
    drow[nbase + 16] = acc1[v];
    drow[nbase + 32] = acc2[v];
    drow[nbase + 48] = acc3[v];
  }
}

// ---------------------------------------------------------------------------
// Skinny exact-fp32 GEMM for xp: N = 33 (masked to 48). 4 waves stacked in M.
// grid = (1, M/64), block = 128
// ---------------------------------------------------------------------------
__global__ __launch_bounds__(128)
void gemm_wmma_xp(const float* __restrict__ A, const float* __restrict__ W,
                  float* __restrict__ D, int K, int ldA, int ldW, int ldD,
                  int Nvalid) {
  const int lane  = threadIdx.x & 31;
  const int wave  = threadIdx.x >> 5;
  const int l15   = lane & 15;
  const int khalf = (lane >> 4) * 2;

  const int arow = blockIdx.y * 64 + wave * 16 + l15;

  v8f acc0 = {}, acc1 = {}, acc2 = {};

  const bool ok1 = (16 + l15) < Nvalid;
  const bool ok2 = (32 + l15) < Nvalid;
  const float* __restrict__ aptr = A + (size_t)arow * ldA + khalf;
  const float* __restrict__ w0 = W + (size_t)(l15 +  0) * ldW + khalf;
  const float* __restrict__ w1 = W + (size_t)(ok1 ? (l15 + 16) : 0) * ldW + khalf;
  const float* __restrict__ w2 = W + (size_t)(ok2 ? (l15 + 32) : 0) * ldW + khalf;

  for (int k0 = 0; k0 < K; k0 += 4) {
    v2f a  = *(const v2f*)(aptr + k0);
    v2f b0 = *(const v2f*)(w0 + k0);
    v2f b1 = *(const v2f*)(w1 + k0);
    v2f b2 = *(const v2f*)(w2 + k0);
    if (!ok1) { b1.x = 0.f; b1.y = 0.f; }
    if (!ok2) { b2.x = 0.f; b2.y = 0.f; }
    acc0 = wmma_f32(a, b0, acc0);
    acc1 = wmma_f32(a, b1, acc1);
    acc2 = wmma_f32(a, b2, acc2);
  }

  const int srow0 = blockIdx.y * 64 + wave * 16 + (lane >> 4) * 8;
  #pragma unroll
  for (int v = 0; v < 8; ++v) {
    float* drow = D + (size_t)(srow0 + v) * ldD;
    drow[l15] = acc0[v];
    if (ok1) drow[l15 + 16] = acc1[v];
    if (ok2) drow[l15 + 32] = acc2[v];
  }
}

// ---------------------------------------------------------------------------
// Depthwise causal conv1d (k=4) + bias + SiLU on xc half of xz.
// ---------------------------------------------------------------------------
__global__ __launch_bounds__(256)
void conv_silu_kernel(const float* __restrict__ xz, const float* __restrict__ cw,
                      const float* __restrict__ cb, float* __restrict__ xc_act) {
  const int idx = blockIdx.x * 256 + threadIdx.x;          // over B*L*DINNER
  if (idx >= BATCH * LSEQ * DINNER) return;
  const int d  = idx & (DINNER - 1);
  const int bl = idx >> 11;                                // b*LSEQ + l
  const int l  = bl & (LSEQ - 1);
  const int b  = bl >> 10;

  float acc = cb[d];
  #pragma unroll
  for (int j = 0; j < 4; ++j) {
    const int ls = l - 3 + j;
    if (ls >= 0)
      acc += cw[d * 4 + j] * xz[(size_t)(b * LSEQ + ls) * (2 * DINNER) + d];
  }
  const float s = acc / (1.f + __expf(-acc));              // SiLU
  xc_act[(size_t)bl * DINNER + d] = s;
}

// ---------------------------------------------------------------------------
// Fused selective scan: one thread per (b, d) channel, h[16] in registers.
// Per-step xp[b,l,0:33] broadcast via LDS. Fuses delta=softplus, dA=exp,
// D-skip connection and silu(z) gating.  grid = (DINNER/256, B)
// ---------------------------------------------------------------------------
__global__ __launch_bounds__(256)
void scan_kernel(const float* __restrict__ xz, const float* __restrict__ xc,
                 const float* __restrict__ xp, const float* __restrict__ w_dt,
                 const float* __restrict__ b_dt, const float* __restrict__ A_log,
                 const float* __restrict__ D_param, float* __restrict__ y) {
  const int d = blockIdx.x * 256 + threadIdx.x;
  const int b = blockIdx.y;
  __shared__ float s_xp[NXP];

  float h[DSTATE], An[DSTATE];
  #pragma unroll
  for (int n = 0; n < DSTATE; ++n) {
    h[n]  = 0.f;
    An[n] = -__expf(A_log[d * DSTATE + n]);
  }
  const float wdt = w_dt[d];
  const float bdt = b_dt[d];
  const float Dp  = D_param[d];

  for (int l = 0; l < LSEQ; ++l) {
    __syncthreads();
    if (threadIdx.x < NXP)
      s_xp[threadIdx.x] = xp[(size_t)(b * LSEQ + l) * NXP + threadIdx.x];
    __syncthreads();

    const size_t bl = (size_t)b * LSEQ + l;
    const float xcv = xc[bl * DINNER + d];
    const float raw = s_xp[2 * DSTATE] * wdt + bdt;
    const float delta = (raw > 20.f) ? raw : log1pf(__expf(raw));  // softplus
    const float dxc = delta * xcv;

    float yacc = 0.f;
    #pragma unroll
    for (int n = 0; n < DSTATE; ++n) {
      const float dA = __expf(delta * An[n]);
      h[n] = dA * h[n] + dxc * s_xp[n];
      yacc += h[n] * s_xp[DSTATE + n];
    }

    const float zv = xz[bl * (2 * DINNER) + DINNER + d];
    const float gate = zv / (1.f + __expf(-zv));                   // silu(z)
    y[bl * DINNER + d] = (yacc + Dp * xcv) * gate;
  }
}

// ---------------------------------------------------------------------------
extern "C" void kernel_launch(void* const* d_in, const int* in_sizes, int n_in,
                              void* d_out, int out_size, void* d_ws, size_t ws_size,
                              hipStream_t stream) {
  const float* x       = (const float*)d_in[0];  // (B, L, 1024)
  const float* W_in    = (const float*)d_in[1];  // (4096, 1024)
  const float* conv_w  = (const float*)d_in[2];  // (2048, 1, 4)
  const float* conv_b  = (const float*)d_in[3];  // (2048,)
  const float* W_x     = (const float*)d_in[4];  // (33, 2048)
  const float* w_dt    = (const float*)d_in[5];  // (2048,)
  const float* b_dt    = (const float*)d_in[6];  // (2048,)
  const float* A_log   = (const float*)d_in[7];  // (2048, 16)
  const float* D_param = (const float*)d_in[8];  // (2048,)
  const float* W_out   = (const float*)d_in[9];  // (1024, 2048)
  float* out = (float*)d_out;                    // (B, L, 1024)

  char* wsb = (char*)d_ws;
  const size_t MB = 1u << 20;
  float* xz = (float*)(wsb);                         // 32 MB
  float* xc = (float*)(wsb + 32 * MB);               // 16 MB
  float* xp = (float*)(wsb + 48 * MB);               // 0.5 MB (padded)
  float* y  = (float*)(wsb + 48 * MB + MB / 2);      // 16 MB
  // activation split region (x: 4+4 MB; later y: 8+8 MB overlays)
  unsigned short* act_hi = (unsigned short*)(wsb + 65 * MB);       // 8 MB slot
  unsigned short* act_lo = (unsigned short*)(wsb + 73 * MB);       // 8 MB slot
  // weight split region (W_in: 8+8 MB; later W_out: 4+4 MB overlays)
  unsigned short* w_hi   = (unsigned short*)(wsb + 81 * MB);       // 8 MB slot
  unsigned short* w_lo   = (unsigned short*)(wsb + 89 * MB);       // 8 MB slot

  const int n_x   = M_ROWS * DMODEL;        // 2,097,152
  const int n_win = 2 * DINNER * DMODEL;    // 4,194,304
  const int n_y   = M_ROWS * DINNER;        // 4,194,304
  const int n_wo  = DMODEL * DINNER;        // 2,097,152

  // 1) split x and W_in to bf16 hi/lo
  split_bf16_kernel<<<(n_x + 255) / 256, 256, 0, stream>>>(x, act_hi, act_lo, n_x);
  split_bf16_kernel<<<(n_win + 255) / 256, 256, 0, stream>>>(W_in, w_hi, w_lo, n_win);

  // 2) xz = x @ W_in^T   (M=2048, K=1024, N=4096), bf16x3
  gemm_bf16x3<<<dim3(4096 / 128, M_ROWS / 32), 128, 0, stream>>>(
      act_hi, act_lo, w_hi, w_lo, xz, DMODEL, 2 * DINNER);

  // 3) depthwise conv + bias + SiLU -> xc
  conv_silu_kernel<<<(BATCH * LSEQ * DINNER) / 256, 256, 0, stream>>>(
      xz, conv_w, conv_b, xc);

  // 4) xp = xc @ W_x^T   (M=2048, K=2048, N=33), exact fp32 WMMA
  gemm_wmma_xp<<<dim3(1, M_ROWS / 64), 128, 0, stream>>>(
      xc, W_x, xp, DINNER, DINNER, DINNER, NXP, NXP);

  // 5) fused selective scan + skip + gating -> y
  scan_kernel<<<dim3(DINNER / 256, BATCH), 256, 0, stream>>>(
      xz, xc, xp, w_dt, b_dt, A_log, D_param, y);

  // 6) split y and W_out to bf16 hi/lo (reuse regions)
  split_bf16_kernel<<<(n_y + 255) / 256, 256, 0, stream>>>(y, act_hi, act_lo, n_y);
  split_bf16_kernel<<<(n_wo + 255) / 256, 256, 0, stream>>>(W_out, w_hi, w_lo, n_wo);

  // 7) out = y @ W_out^T (M=2048, K=2048, N=1024), bf16x3
  gemm_bf16x3<<<dim3(1024 / 128, M_ROWS / 32), 128, 0, stream>>>(
      act_hi, act_lo, w_hi, w_lo, out, DINNER, DMODEL);
}
